// DiscriminativeLSTM_35682588295846
// MI455X (gfx1250) — matrix operand order repaired and checked by reading, not compiled, not run on hardware
//
#include <hip/hip_runtime.h>

typedef __attribute__((ext_vector_type(16))) _Float16 v16h;
typedef __attribute__((ext_vector_type(8)))  _Float16 v8h;
typedef __attribute__((ext_vector_type(8)))  float    v8f;

#define HDIM 64
#define TSTEPS 1024
#define NTILES 16         // 4H / 16
#define KC 2              // K=64 -> two K-chunks of 32
#define PACK_ELEMS (NTILES * KC * 32 * 16)   // 16384 f16 per matrix
#define LDS_STRIDE 72     // padded f16 row stride (144B, 16B-aligned rows)
#define HBUF (16 * LDS_STRIDE)

// ---------------------------------------------------------------------------
// Pack a [256 x 64] f32 weight matrix (row = gate n, col = k) into the CDNA5
// WMMA B-matrix (32x16 f16) lane layout:
//   flat = ((tile*KC + kk)*32 + lane)*16 + e
//   value = W[n*64 + k],  n = tile*16 + (lane&15),  k = kk*32 + (e>=8?16:0) + (lane>=16?8:0) + (e&7)
// ---------------------------------------------------------------------------
__global__ void pack_b_kernel(const float* __restrict__ Whh0,
                              const float* __restrict__ Wih1,
                              const float* __restrict__ Whh1,
                              _Float16* __restrict__ out) {
  int idx = blockIdx.x * blockDim.x + threadIdx.x;
  if (idx >= 3 * PACK_ELEMS) return;
  int mat = idx / PACK_ELEMS;
  int r   = idx % PACK_ELEMS;
  int e    = r & 15;
  int lane = (r >> 4) & 31;
  int kk   = (r >> 9) & 1;
  int tile = r >> 10;
  int n = tile * 16 + (lane & 15);
  int k = kk * 32 + ((e >> 3) << 4) + ((lane >> 4) << 3) + (e & 7);
  const float* W = (mat == 0) ? Whh0 : (mat == 1 ? Wih1 : Whh1);
  out[idx] = (_Float16)W[n * HDIM + k];
}

// Load one 16x32 f16 A operand from an LDS row-major [16][LDS_STRIDE] buffer.
// Lane L: M = L&15; K runs = kk*32 + (L>=16?8:0) + {0..7, 16..23}.
__device__ inline v16h load_a(const _Float16* lds, int mrow, int kk, int hi) {
  int k0 = kk * 32 + hi * 8;
  const v8h* p0 = (const v8h*)(lds + mrow * LDS_STRIDE + k0);
  const v8h* p1 = (const v8h*)(lds + mrow * LDS_STRIDE + k0 + 16);
  v8h lo = *p0;
  v8h hh = *p1;
  v16h a;
#pragma unroll
  for (int i = 0; i < 8; ++i) { a[i] = lo[i]; a[8 + i] = hh[i]; }
  return a;
}

// Load one packed 32x16 f16 B operand (32B per lane).
__device__ inline v16h load_b(const _Float16* __restrict__ pk, int tile, int kk, int lane) {
  const v16h* p = (const v16h*)(pk + (size_t)((tile * KC + kk) * 32 + lane) * 16);
  return *p;
}

__device__ inline float sigm(float v) {
  return __builtin_amdgcn_rcpf(1.0f + __expf(-v));   // v_exp + v_add + v_rcp
}
__device__ inline float tanh_fast(float v) {
  float e = __expf(-2.0f * v);
  return (1.0f - e) * __builtin_amdgcn_rcpf(1.0f + e);
}

__launch_bounds__(128)
__global__ void lstm2_wmma_kernel(const float* __restrict__ x,
                                  const float* __restrict__ Wih0,
                                  const float* __restrict__ bih0,
                                  const float* __restrict__ bhh0,
                                  const float* __restrict__ bih1,
                                  const float* __restrict__ bhh1,
                                  const float* __restrict__ Wc1,
                                  const float* __restrict__ bc1,
                                  const float* __restrict__ Wc2,
                                  const float* __restrict__ bc2,
                                  const _Float16* __restrict__ pack,
                                  float* __restrict__ out) {
  // 4 h buffers: h1 ping/pong, h2 ping/pong (f16 row-major [16][LDS_STRIDE])
  __shared__ _Float16 lds[4 * HBUF];

  const int tid  = threadIdx.x;
  const int lane = tid & 31;           // lane in wave32
  const int wv   = tid >> 5;           // wave id = gate-column subgroup j (0..3)
  const int nl   = lane & 15;
  const int hi   = lane >> 4;
  const int b0   = blockIdx.x * 16;    // batch tile base

  for (int i = tid; i < 4 * HBUF; i += 128) lds[i] = (_Float16)0.0f;

  // ---- register-resident B operands for this wave's 4 gate tiles ----
  // tiles tn = g*4 + wv for g = 0..3 (i, f, g, o columns 16*wv..16*wv+15)
  v16h bR0[4][2], bI1[4][2], bR1[4][2];
#pragma unroll
  for (int g = 0; g < 4; ++g) {
    int tn = g * 4 + wv;
#pragma unroll
    for (int kk = 0; kk < 2; ++kk) {
      bR0[g][kk] = load_b(pack + 0 * PACK_ELEMS, tn, kk, lane);   // Whh0
      bI1[g][kk] = load_b(pack + 1 * PACK_ELEMS, tn, kk, lane);   // Wih1
      bR1[g][kk] = load_b(pack + 2 * PACK_ELEMS, tn, kk, lane);   // Whh1
    }
  }

  float wi0[4], bs0[4], bs1[4];
#pragma unroll
  for (int g = 0; g < 4; ++g) {
    int n = (g * 4 + wv) * 16 + nl;
    wi0[g] = Wih0[n];                  // D == 1
    bs0[g] = bih0[n] + bhh0[n];
    bs1[g] = bih1[n] + bhh1[n];
  }

  // cell state for this wave's 16 hidden columns (rows r + hi*8, col wv*16+nl)
  v8f c1, c2;
#pragma unroll
  for (int r = 0; r < 8; ++r) { c1[r] = 0.0f; c2[r] = 0.0f; }

  const _Float16* Ard = lds;                 // h1_{t-1}
  _Float16*       Awr = lds + HBUF;          // h1_t
  const _Float16* Brd = lds + 2 * HBUF;      // h2_{t-1}
  _Float16*       Bwr = lds + 3 * HBUF;      // h2_t
  __syncthreads();

  for (int t = 0; t < TSTEPS; ++t) {
    // ---------------- layer 0 ----------------
    v16h aA0 = load_a(Ard, nl, 0, hi);
    v16h aA1 = load_a(Ard, nl, 1, hi);

    float xv[8];
#pragma unroll
    for (int r = 0; r < 8; ++r)
      xv[r] = x[(size_t)(b0 + r + hi * 8) * TSTEPS + t];

    v8f acc[4];
#pragma unroll
    for (int g = 0; g < 4; ++g) {
      v8f c;
#pragma unroll
      for (int r = 0; r < 8; ++r) c[r] = fmaf(xv[r], wi0[g], bs0[g]);
      c = __builtin_amdgcn_wmma_f32_16x16x32_f16(false, aA0, false, bR0[g][0],
                                                 (short)0, c, false, false);
      c = __builtin_amdgcn_wmma_f32_16x16x32_f16(false, aA1, false, bR0[g][1],
                                                 (short)0, c, false, false);
      acc[g] = c;
    }
#pragma unroll
    for (int r = 0; r < 8; ++r) {
      float iv = sigm(acc[0][r]);
      float fv = sigm(acc[1][r]);
      float gv = tanh_fast(acc[2][r]);
      float ov = sigm(acc[3][r]);
      float cn = fmaf(fv, c1[r], iv * gv);
      c1[r] = cn;
      float hv = ov * tanh_fast(cn);
      Awr[(r + hi * 8) * LDS_STRIDE + wv * 16 + nl] = (_Float16)hv;
    }
    __syncthreads();   // h1_t complete & visible; everyone done reading Ard

    // ---------------- layer 1 ----------------
    v16h a10 = load_a(Awr, nl, 0, hi);
    v16h a11 = load_a(Awr, nl, 1, hi);
    v16h aB0 = load_a(Brd, nl, 0, hi);
    v16h aB1 = load_a(Brd, nl, 1, hi);

#pragma unroll
    for (int g = 0; g < 4; ++g) {
      v8f c;
#pragma unroll
      for (int r = 0; r < 8; ++r) c[r] = bs1[g];
      c = __builtin_amdgcn_wmma_f32_16x16x32_f16(false, a10, false, bI1[g][0],
                                                 (short)0, c, false, false);
      c = __builtin_amdgcn_wmma_f32_16x16x32_f16(false, a11, false, bI1[g][1],
                                                 (short)0, c, false, false);
      c = __builtin_amdgcn_wmma_f32_16x16x32_f16(false, aB0, false, bR1[g][0],
                                                 (short)0, c, false, false);
      c = __builtin_amdgcn_wmma_f32_16x16x32_f16(false, aB1, false, bR1[g][1],
                                                 (short)0, c, false, false);
      acc[g] = c;
    }
#pragma unroll
    for (int r = 0; r < 8; ++r) {
      float iv = sigm(acc[0][r]);
      float fv = sigm(acc[1][r]);
      float gv = tanh_fast(acc[2][r]);
      float ov = sigm(acc[3][r]);
      float cn = fmaf(fv, c2[r], iv * gv);
      c2[r] = cn;
      float hv = ov * tanh_fast(cn);
      Bwr[(r + hi * 8) * LDS_STRIDE + wv * 16 + nl] = (_Float16)hv;
    }
    __syncthreads();   // h2_t complete; everyone done reading Brd

    // swap ping-pong buffers
    const _Float16* tA = Ard; Ard = Awr; Awr = (_Float16*)tA;
    const _Float16* tB = Brd; Brd = Bwr; Bwr = (_Float16*)tB;
  }

  // ---- classifier head on final h2 (in Brd after last swap) ----
  if (wv == 0 && lane < 16) {
    float acc2 = bc2[0];
    for (int j = 0; j < 32; ++j) {
      float a = bc1[j];
      for (int k = 0; k < HDIM; ++k)
        a = fmaf((float)Brd[lane * LDS_STRIDE + k], Wc1[j * HDIM + k], a);
      a = fmaxf(a, 0.0f);
      acc2 = fmaf(a, Wc2[j], acc2);
    }
    out[b0 + lane] = acc2;
  }
}

extern "C" void kernel_launch(void* const* d_in, const int* in_sizes, int n_in,
                              void* d_out, int out_size, void* d_ws, size_t ws_size,
                              hipStream_t stream) {
  const float* x    = (const float*)d_in[0];
  const float* Wih0 = (const float*)d_in[1];
  const float* Whh0 = (const float*)d_in[2];
  const float* bih0 = (const float*)d_in[3];
  const float* bhh0 = (const float*)d_in[4];
  const float* Wih1 = (const float*)d_in[5];
  const float* Whh1 = (const float*)d_in[6];
  const float* bih1 = (const float*)d_in[7];
  const float* bhh1 = (const float*)d_in[8];
  const float* Wc1  = (const float*)d_in[9];
  const float* bc1  = (const float*)d_in[10];
  const float* Wc2  = (const float*)d_in[11];
  const float* bc2  = (const float*)d_in[12];

  _Float16* pack = (_Float16*)d_ws;

  int tot = 3 * PACK_ELEMS;
  pack_b_kernel<<<(tot + 255) / 256, 256, 0, stream>>>(Whh0, Wih1, Whh1, pack);

  int B = in_sizes[0] / TSTEPS;   // x is [B, T, 1]
  lstm2_wmma_kernel<<<B / 16, 128, 0, stream>>>(x, Wih0, bih0, bhh0, bih1, bhh1,
                                                Wc1, bc1, Wc2, bc2, pack,
                                                (float*)d_out);
}
